// StyleEncoder_37340445671528
// MI455X (gfx1250) — matrix operand order repaired and checked by reading, not compiled, not run on hardware
//
#include <hip/hip_runtime.h>
#include <stdint.h>

// ---------------------------------------------------------------------------
// Types
// ---------------------------------------------------------------------------
typedef __attribute__((ext_vector_type(16))) __bf16 bf16x16;
typedef __attribute__((ext_vector_type(8)))  float  f32x8;

struct alignas(16) U128 { unsigned int x[4]; };
struct alignas(8)  U64  { unsigned int x[2]; };

union FragU { U128 q[2]; bf16x16 v; };

__device__ __forceinline__ unsigned short f2bf(float f) {
  union { float f; unsigned int u; } x; x.f = f;
  unsigned int r = x.u + 0x7fffu + ((x.u >> 16) & 1u);  // round-to-nearest-even
  return (unsigned short)(r >> 16);
}
__device__ __forceinline__ float bf2f(unsigned short b) {
  union { unsigned int u; float f; } x; x.u = ((unsigned int)b) << 16;
  return x.f;
}

#define NUM_INST 20

// input coordinate calc; mode 0: plain conv, 1: lhs-dilated deconv (pad 1,2),
// 2: reflect-pad stride-1 conv
__device__ __forceinline__ bool calc_coords(int mode, int oh, int ow, int kh, int kw,
                                            int stride, int pad, int Hin, int Win,
                                            int& ih, int& iw) {
  if (mode == 0) {
    ih = oh * stride - pad + kh;
    iw = ow * stride - pad + kw;
    return ((unsigned)ih < (unsigned)Hin) & ((unsigned)iw < (unsigned)Win);
  } else if (mode == 1) {
    int ph = oh - 1 + kh, pw = ow - 1 + kw;
    bool ok = (ph >= 0) & (pw >= 0) & ((ph & 1) == 0) & ((pw & 1) == 0);
    ih = ph >> 1; iw = pw >> 1;
    return ok & (ih < Hin) & (iw < Win);
  } else {
    int iy = oh - pad + kh, ix = ow - pad + kw;
    ih = (iy < 0) ? -iy : ((iy >= Hin) ? 2 * Hin - 2 - iy : iy);
    iw = (ix < 0) ? -ix : ((ix >= Win) ? 2 * Win - 2 - ix : ix);
    return true;
  }
}

// ---------------------------------------------------------------------------
// Implicit-GEMM conv, bf16 WMMA, 2x2 register tiling per wave, register-
// prefetch software pipelining (next K-tile loaded to VGPRs during WMMA).
//   M = B*Hout*Wout, N = Cout, K = Kh*Kw*Cin.
//   Weights pre-transposed [Cout][Ktot] bf16 -> straight b128/b64 staging.
//   LDS tiles: A [MTILE][32], B(n-major) [NTILE][32], row stride 40 halves.
//   Fragments use k = 16*laneHi + s (not ISA table order): each fragment == 2 ds_load_b128
//   (K is contiguous per half-lane).
// ---------------------------------------------------------------------------
#define LDS_S 40

template<int MTILE, int NTILE, bool FASTA>
__global__ __launch_bounds__(256) void conv_wmma_kernel(
    const unsigned short* __restrict__ in,    // [B,Hin,Win,Cin] bf16 bits
    const unsigned short* __restrict__ wgt,   // [Cout][Ktot] bf16 bits
    const float* __restrict__ bias,
    unsigned short* __restrict__ out_bf,      // [M,Cout] or nullptr
    float* __restrict__ out_f32,              // tanh f32 [M,Cout] or nullptr
    float* __restrict__ stat_sum,             // [Cout] or nullptr
    float* __restrict__ stat_sq,
    int Hin, int Win, int Cin,
    int Hout, int Wout, int Cout,
    int Kh, int Kw, int stride, int pad, int mode, int Ktot,
    int wShift, int cinShift, int magicCin, int magicKw)
{
  constexpr int nWaves = NTILE / 32;           // 1 or 2
  constexpr int mWaves = 8 / nWaves;           // 8 or 4
  static_assert(MTILE == mWaves * 32, "tile/wave mismatch");
  constexpr int EPT = MTILE / 8;               // A halves per thread (16 or 32)
  constexpr int TPR = 32 / EPT;                // threads per A row
  constexpr int BPT = NTILE / 8;               // B halves per thread (8 or 4)
  constexpr int TPN = 32 / BPT;                // threads per B row
  constexpr int ACH = EPT / 8;                 // A b128 chunks per thread

  __shared__ unsigned short lA[MTILE * LDS_S];
  __shared__ unsigned short lB[NTILE * LDS_S];

  const int tid    = threadIdx.x;
  const int lane   = tid & 31;
  const int wid    = tid >> 5;
  const int waveM  = wid % mWaves;
  const int waveN  = wid / mWaves;
  const int laneLo = lane & 15;
  const int laneHi = lane >> 4;
  const int mWaveBase = waveM * 32;
  const int nWaveBase = waveN * 32;

  const int mBlock = blockIdx.x * MTILE;
  const int nBlock = blockIdx.y * NTILE;

  // A loader coords (per thread: one row, EPT consecutive k)
  const int aRow = tid / TPR;
  const int aK   = (tid % TPR) * EPT;
  const int mG = mBlock + aRow;
  const int ow = mG & (Wout - 1);
  const int oh = (mG >> wShift) & (Hout - 1);
  const int bb = mG >> (2 * wShift);           // Hout == Wout (square)

  // B loader coords
  const int bN = tid / TPN;
  const int bK = (tid % TPN) * BPT;
  const int nG = nBlock + bN;

  // prefetch registers
  union APre { U128 q[ACH]; unsigned int w[EPT / 2]; };
  union BPre { U128 q; U64 d; };
  APre aPre;
  BPre bPre;

  auto preloadA = [&](int k0) {
    if (FASTA) {
      // Cin pow2 >= 16, Ktot % 32 == 0: 8 consecutive k == 8 consecutive ci
      #pragma unroll
      for (int jc = 0; jc < ACH; ++jc) {
        int k  = k0 + aK + jc * 8;
        int ci = k & (Cin - 1);
        int r  = k >> cinShift;
        int kh = (r * magicKw) >> 16;
        int kw = r - kh * Kw;
        int ih, iw;
        bool ok = calc_coords(mode, oh, ow, kh, kw, stride, pad, Hin, Win, ih, iw);
        U128 v = {};
        if (ok) v = *(const U128*)&in[((bb * Hin + ih) * Win + iw) * Cin + ci];
        aPre.q[jc] = v;
      }
    } else {
      // scalar path (stem, Cin==3) with incremental (ci,kw,kh) decode
      int k  = k0 + aK;
      int r  = (k * magicCin) >> 16;
      int ci = k - r * Cin;
      int kh = (r * magicKw) >> 16;
      int kw = r - kh * Kw;
      #pragma unroll
      for (int j = 0; j < EPT; ++j) {
        unsigned short v = 0;
        if (k + j < Ktot) {
          int ih, iw;
          bool ok = calc_coords(mode, oh, ow, kh, kw, stride, pad, Hin, Win, ih, iw);
          if (ok) v = in[((bb * Hin + ih) * Win + iw) * Cin + ci];
        }
        if ((j & 1) == 0) aPre.w[j >> 1] = v;
        else              aPre.w[j >> 1] |= ((unsigned int)v) << 16;
        ++ci;
        if (ci == Cin) { ci = 0; ++kw; if (kw == Kw) { kw = 0; ++kh; } }
      }
    }
  };

  auto preloadB = [&](int k0) {
    if (FASTA) {
      if (BPT == 8) {
        U128 v = {};
        if (nG < Cout) v = *(const U128*)&wgt[nG * Ktot + k0 + bK];
        bPre.q = v;
      } else {
        U64 v = {};
        if (nG < Cout) v = *(const U64*)&wgt[nG * Ktot + k0 + bK];
        bPre.d = v;
      }
    } else {
      U128 v = {};
      unsigned short* h = (unsigned short*)&v;
      #pragma unroll
      for (int j = 0; j < BPT; ++j) {
        int k = k0 + bK + j;
        if (nG < Cout && k < Ktot) h[j] = wgt[nG * Ktot + k];
      }
      bPre.q = v;
    }
  };

  f32x8 acc[2][2];
  #pragma unroll
  for (int mi = 0; mi < 2; ++mi)
    #pragma unroll
    for (int ni = 0; ni < 2; ++ni)
      acc[mi][ni] = (f32x8){0.f,0.f,0.f,0.f,0.f,0.f,0.f,0.f};

  preloadA(0);
  preloadB(0);

  for (int k0 = 0; k0 < Ktot; k0 += 32) {
    // ---- commit prefetched tile to LDS ----
    #pragma unroll
    for (int jc = 0; jc < ACH; ++jc)
      *(U128*)&lA[aRow * LDS_S + aK + jc * 8] = aPre.q[jc];
    if (BPT == 8) *(U128*)&lB[bN * LDS_S + bK] = bPre.q;
    else          *(U64*) &lB[bN * LDS_S + bK] = bPre.d;
    __syncthreads();

    // ---- prefetch next K-tile (overlaps fragment reads + WMMA below) ----
    int k1 = k0 + 32;
    if (k1 < Ktot) { preloadA(k1); preloadB(k1); }

    // ---- fragments: two ds_load_b128 each ----
    FragU fa[2], fb[2];
    #pragma unroll
    for (int mi = 0; mi < 2; ++mi) {
      int row = mWaveBase + mi * 16 + laneLo;
      fa[mi].q[0] = *(const U128*)&lA[row * LDS_S + 16 * laneHi];
      fa[mi].q[1] = *(const U128*)&lA[row * LDS_S + 16 * laneHi + 8];
    }
    #pragma unroll
    for (int ni = 0; ni < 2; ++ni) {
      int row = nWaveBase + ni * 16 + laneLo;
      fb[ni].q[0] = *(const U128*)&lB[row * LDS_S + 16 * laneHi];
      fb[ni].q[1] = *(const U128*)&lB[row * LDS_S + 16 * laneHi + 8];
    }
    #pragma unroll
    for (int mi = 0; mi < 2; ++mi)
      #pragma unroll
      for (int ni = 0; ni < 2; ++ni)
        acc[mi][ni] = __builtin_amdgcn_wmma_f32_16x16x32_bf16(
            false, fa[mi].v, false, fb[ni].v, (short)0, acc[mi][ni], false, false);
    __syncthreads();
  }

  // ---------------- epilogue ----------------
  #pragma unroll
  for (int ni = 0; ni < 2; ++ni) {
    int cG = nBlock + nWaveBase + ni * 16 + laneLo;
    float bv = (bias && cG < Cout) ? bias[cG] : 0.f;
    float s = 0.f, s2 = 0.f;
    #pragma unroll
    for (int mi = 0; mi < 2; ++mi) {
      int mBase = mBlock + mWaveBase + mi * 16 + laneHi * 8;
      #pragma unroll
      for (int r = 0; r < 8; ++r) {
        float val = acc[mi][ni][r] + bv;
        s += val; s2 += val * val;
        if (cG < Cout) {
          int mRow = mBase + r;
          if (out_bf)  out_bf[mRow * Cout + cG] = f2bf(val);
          if (out_f32) out_f32[mRow * Cout + cG] = tanhf(val);
        }
      }
    }
    if (stat_sum) {  // BN batch statistics
      s  += __shfl_xor(s, 16);
      s2 += __shfl_xor(s2, 16);
      if (laneHi == 0 && cG < Cout) {
        atomicAdd(&stat_sum[cG], s);
        atomicAdd(&stat_sq[cG], s2);
      }
    }
  }
}

// ---------------------------------------------------------------------------
// Elementwise helpers
// ---------------------------------------------------------------------------
__global__ void nchw2nhwc_bf16(const float* __restrict__ in,
                               unsigned short* __restrict__ out,
                               int C, int HW, int total) {
  int i = blockIdx.x * blockDim.x + threadIdx.x;
  if (i >= total) return;
  int c = i % C;
  int sIdx = i / C;
  int hw = sIdx % HW;
  int b  = sIdx / HW;
  out[i] = f2bf(in[(b * C + c) * HW + hw]);
}

// OIHW f32 -> transposed [Cout][K=(kh*Kw+kw)*Cin+ci] bf16
__global__ void repack_w(const float* __restrict__ w, unsigned short* __restrict__ o,
                         int Cin, int Cout, int Kh, int Kw, int Ktot, int total) {
  int e = blockIdx.x * blockDim.x + threadIdx.x;
  if (e >= total) return;
  int k  = e % Ktot;
  int n  = e / Ktot;
  int ci = k % Cin;
  int r  = k / Cin;
  int kw = r % Kw;
  int kh = r / Kw;
  o[e] = f2bf(w[((n * Cin + ci) * Kh + kh) * Kw + kw]);
}

__global__ void bn_relu_kernel(const unsigned short* __restrict__ x,
                               unsigned short* __restrict__ y,
                               const float* __restrict__ ssum,
                               const float* __restrict__ ssq,
                               const float* __restrict__ gamma,
                               const float* __restrict__ beta,
                               int total, int cMask, float invM) {
  int i = blockIdx.x * blockDim.x + threadIdx.x;
  if (i >= total) return;
  int c = i & cMask;   // C is pow2 for all BN layers
  float mu  = ssum[c] * invM;
  float var = ssq[c] * invM - mu * mu;
  float v = (bf2f(x[i]) - mu) * rsqrtf(var + 1e-5f) * gamma[c] + beta[c];
  y[i] = f2bf(fmaxf(v, 0.f));
}

// ---------------------------------------------------------------------------
// Instance-mean
// ---------------------------------------------------------------------------
__global__ __launch_bounds__(256) void seg_scatter(const float* __restrict__ feat, // [P,3]
                                                   const int* __restrict__ inst,
                                                   float* __restrict__ segsum,     // [20*3]
                                                   float* __restrict__ segcnt,     // [20]
                                                   int P) {
  __shared__ float ls[NUM_INST * 4];
  for (int i = threadIdx.x; i < NUM_INST * 4; i += 256) ls[i] = 0.f;
  __syncthreads();
  for (int p = blockIdx.x * 256 + threadIdx.x; p < P; p += gridDim.x * 256) {
    int id = inst[p];
    atomicAdd(&ls[id * 4 + 0], feat[p * 3 + 0]);
    atomicAdd(&ls[id * 4 + 1], feat[p * 3 + 1]);
    atomicAdd(&ls[id * 4 + 2], feat[p * 3 + 2]);
    atomicAdd(&ls[id * 4 + 3], 1.f);
  }
  __syncthreads();
  for (int i = threadIdx.x; i < NUM_INST * 4; i += 256) {
    int id = i >> 2, c = i & 3;
    float v = ls[i];
    if (c < 3) atomicAdd(&segsum[id * 3 + c], v);
    else       atomicAdd(&segcnt[id], v);
  }
}

__global__ void seg_mean(float* __restrict__ segsum, const float* __restrict__ segcnt) {
  int i = threadIdx.x;
  if (i < NUM_INST * 3) segsum[i] = segsum[i] / fmaxf(segcnt[i / 3], 1.f);
}

__global__ void seg_gather(const float* __restrict__ means, const int* __restrict__ inst,
                           float* __restrict__ out, int H, int W, int total) {
  int i = blockIdx.x * blockDim.x + threadIdx.x;
  if (i >= total) return;
  int w = i % W; int t = i / W;
  int h = t % H; t /= H;
  int c = t % 3; int b = t / 3;
  int id = inst[(b * H + h) * W + w];
  out[i] = means[id * 3 + c];
}

// ---------------------------------------------------------------------------
// Host orchestration
// ---------------------------------------------------------------------------
struct Spec { int Cin, Cout, K, stride, pad, mode, Hin, Hout, wi; };

static inline int ilog2i(int v) { int s = 0; while ((1 << s) < v) ++s; return s; }

extern "C" void kernel_launch(void* const* d_in, const int* in_sizes, int n_in,
                              void* d_out, int out_size, void* d_ws, size_t ws_size,
                              hipStream_t stream) {
  (void)in_sizes; (void)n_in; (void)out_size; (void)ws_size;
  const float* real_img = (const float*)d_in[0];
  const int*   inst     = (const int*)d_in[1];

  static const Spec sp[10] = {
      {  3,  32, 7, 1, 3, 2, 512, 512,  2},   // stem (reflect)
      { 32,  64, 3, 2, 1, 0, 512, 256,  6},   // down0
      { 64, 128, 3, 2, 1, 0, 256, 128, 10},   // down1
      {128, 256, 3, 2, 1, 0, 128,  64, 14},   // down2
      {256, 512, 3, 2, 1, 0,  64,  32, 18},   // down3
      {512, 256, 3, 1, 1, 1,  32,  64, 22},   // up0 (deconv)
      {256, 128, 3, 1, 1, 1,  64, 128, 26},   // up1
      {128,  64, 3, 1, 1, 1, 128, 256, 30},   // up2
      { 64,  32, 3, 1, 1, 1, 256, 512, 34},   // up3
      { 32,   3, 7, 1, 3, 2, 512, 512, 38},   // head (reflect, tanh)
  };

  // workspace layout
  char* ws = (char*)d_ws;
  size_t off = 0;
  auto take = [&](size_t bytes) { size_t r = off; off += (bytes + 255) & ~size_t(255); return r; };
  const size_t ACT = (size_t)2 * 512 * 512 * 32;
  unsigned short* bufA = (unsigned short*)(ws + take(ACT * 2));
  unsigned short* bufB = (unsigned short*)(ws + take(ACT * 2));
  size_t wTot = 0;
  for (int i = 0; i < 10; ++i) wTot += (size_t)sp[i].K * sp[i].K * sp[i].Cin * sp[i].Cout;
  unsigned short* wbase = (unsigned short*)(ws + take(wTot * 2));
  float* stats = (float*)(ws + take(1024 * sizeof(float)));
  float* feat  = (float*)(ws + take((size_t)2 * 512 * 512 * 3 * 4));
  float* seg   = (float*)(ws + take(80 * sizeof(float)));

  // input NCHW f32 -> NHWC bf16
  int totalIn = 2 * 3 * 512 * 512;
  nchw2nhwc_bf16<<<(totalIn + 255) / 256, 256, 0, stream>>>(real_img, bufB, 3, 512 * 512, totalIn);

  // repack all weights to transposed [Cout][Ktot] bf16
  unsigned short* wptr[10];
  size_t wo = 0;
  for (int i = 0; i < 10; ++i) {
    int Ktot = sp[i].K * sp[i].K * sp[i].Cin;
    int tot  = Ktot * sp[i].Cout;
    wptr[i] = wbase + wo;
    wo += (size_t)tot;
    repack_w<<<(tot + 255) / 256, 256, 0, stream>>>(
        (const float*)d_in[sp[i].wi], wptr[i], sp[i].Cin, sp[i].Cout, sp[i].K, sp[i].K, Ktot, tot);
  }

  auto launch_conv = [&](int i, unsigned short* src, unsigned short* dst_bf,
                         float* dst_f32, float* ss, float* sq) {
    const Spec& s = sp[i];
    int M    = 2 * s.Hout * s.Hout;
    int Ktot = s.K * s.K * s.Cin;
    int wSh  = ilog2i(s.Hout);
    int cSh  = ilog2i(s.Cin);
    int mCin = (65536 + s.Cin - 1) / s.Cin;
    int mKw  = (65536 + s.K - 1) / s.K;
    const float* bptr = (const float*)d_in[s.wi + 1];
    bool fasta = (s.Cin >= 16) && ((s.Cin & (s.Cin - 1)) == 0);
    if (s.Cout >= 64) {
      dim3 grid(M / 128, s.Cout / 64);
      conv_wmma_kernel<128, 64, true><<<grid, 256, 0, stream>>>(
          src, wptr[i], bptr, dst_bf, dst_f32, ss, sq,
          s.Hin, s.Hin, s.Cin, s.Hout, s.Hout, s.Cout,
          s.K, s.K, s.stride, s.pad, s.mode, Ktot, wSh, cSh, mCin, mKw);
    } else if (fasta) {
      dim3 grid(M / 256, 1);
      conv_wmma_kernel<256, 32, true><<<grid, 256, 0, stream>>>(
          src, wptr[i], bptr, dst_bf, dst_f32, ss, sq,
          s.Hin, s.Hin, s.Cin, s.Hout, s.Hout, s.Cout,
          s.K, s.K, s.stride, s.pad, s.mode, Ktot, wSh, cSh, mCin, mKw);
    } else {
      dim3 grid(M / 256, 1);
      conv_wmma_kernel<256, 32, false><<<grid, 256, 0, stream>>>(
          src, wptr[i], bptr, dst_bf, dst_f32, ss, sq,
          s.Hin, s.Hin, s.Cin, s.Hout, s.Hout, s.Cout,
          s.K, s.K, s.stride, s.pad, s.mode, Ktot, wSh, cSh, mCin, mKw);
    }
  };

  // trunk: conv(WMMA, fused BN stats) -> BN+ReLU, ping-pong bufB -> bufA -> bufB
  for (int i = 0; i < 9; ++i) {
    const Spec& s = sp[i];
    int M = 2 * s.Hout * s.Hout;
    hipMemsetAsync(stats, 0, 1024 * sizeof(float), stream);
    launch_conv(i, bufB, bufA, nullptr, stats, stats + 512);
    int tot = M * s.Cout;
    bn_relu_kernel<<<(tot + 255) / 256, 256, 0, stream>>>(
        bufA, bufB, stats, stats + 512,
        (const float*)d_in[s.wi + 2], (const float*)d_in[s.wi + 3],
        tot, s.Cout - 1, 1.0f / (float)M);
  }

  // head conv: reflect pad 3, 7x7, 32->3, tanh, f32 NHWC out
  launch_conv(9, bufB, nullptr, feat, nullptr, nullptr);

  // instance means
  hipMemsetAsync(seg, 0, 80 * sizeof(float), stream);
  int P = 2 * 512 * 512;
  seg_scatter<<<512, 256, 0, stream>>>(feat, inst, seg, seg + 60, P);
  seg_mean<<<1, 64, 0, stream>>>(seg, seg + 60);
  int totalOut = 2 * 3 * 512 * 512;
  seg_gather<<<(totalOut + 255) / 256, 256, 0, stream>>>(
      seg, inst, (float*)d_out, 512, 512, totalOut);
}
